// BKCoreHyperbolicIntegration_87625922773416
// MI455X (gfx1250) — compile-verified, matrix operand order checked by reading, not run on hardware
//
#include <hip/hip_runtime.h>
#include <hip/hip_bf16.h>

// ---- fixed problem geometry from the reference ----
// B=2, S=1024, H=16, D=2048; P = B*S = 2048 positions; rows = B*H*S = 32768.
#define DD     2048
#define PP     2048
#define SS     1024
#define NROWS  32768   // B*H*S
#define ROWS_PER_B 16384  // H*S

typedef _Float16 h16 __attribute__((ext_vector_type(16)));
typedef _Float16 h8  __attribute__((ext_vector_type(8)));
typedef float    f8  __attribute__((ext_vector_type(8)));
typedef float    f4  __attribute__((ext_vector_type(4)));

// ---------------------------------------------------------------------------
// Kernel 1a: partial column sums of Wd [D x D] row-major.
// grid (8 colBlocks, 32 rowBlocks), 256 threads: thread owns one column d,
// sums 64 rows. Short dependency chains (8-way ILP) keep this latency-hidden.
// ---------------------------------------------------------------------------
__global__ __launch_bounds__(256) void bk_colsum_partial(
    const float* __restrict__ Wd, float* __restrict__ partial)
{
  const int d  = blockIdx.x * 256 + threadIdx.x;
  const int e0 = blockIdx.y * 64;
  const float* p = Wd + (size_t)e0 * DD + d;
  float s0 = 0.f, s1 = 0.f, s2 = 0.f, s3 = 0.f;
  float s4 = 0.f, s5 = 0.f, s6 = 0.f, s7 = 0.f;
  for (int e = 0; e < 64; e += 8) {
    s0 += p[(size_t)(e + 0) * DD];
    s1 += p[(size_t)(e + 1) * DD];
    s2 += p[(size_t)(e + 2) * DD];
    s3 += p[(size_t)(e + 3) * DD];
    s4 += p[(size_t)(e + 4) * DD];
    s5 += p[(size_t)(e + 5) * DD];
    s6 += p[(size_t)(e + 6) * DD];
    s7 += p[(size_t)(e + 7) * DD];
  }
  partial[blockIdx.y * DD + d] = ((s0 + s1) + (s2 + s3)) + ((s4 + s5) + (s6 + s7));
}

// ---------------------------------------------------------------------------
// Kernel 1b: wbar[d] = (mean_e Wd[e,d]) * 1024  (stored f16, pre-scaled into
// the f16 normal range; de-scaled after WMMA accumulation).
// Block 8 reduces bd -> bbar (scalar f32).
// ---------------------------------------------------------------------------
__global__ __launch_bounds__(256) void bk_finalize_wbar(
    const float* __restrict__ partial, const float* __restrict__ bd,
    _Float16* __restrict__ wbar, float* __restrict__ bbar)
{
  const int t = threadIdx.x;
  if (blockIdx.x < 8) {
    const int d = blockIdx.x * 256 + t;
    float s = 0.f;
#pragma unroll
    for (int r = 0; r < 32; ++r) s += partial[r * DD + d];
    // mean = s / D; pre-scale by 1024 -> s * (1024/2048)
    wbar[d] = (_Float16)(s * 0.5f);
  } else {
    __shared__ float red[256];
    float s = 0.f;
    for (int i = t; i < DD; i += 256) s += bd[i];
    red[t] = s;
    __syncthreads();
    for (int off = 128; off > 0; off >>= 1) {
      if (t < off) red[t] += red[t + off];
      __syncthreads();
    }
    if (t == 0) bbar[0] = red[0] * (1.0f / (float)DD);
  }
}

// ---------------------------------------------------------------------------
// Kernel 2: gate[p] via WMMA.  he[p] = dot(x[p,:], wbar)/1024 + bbar, then
// G = 1/(he + EPS - 0.1i), gate = sigmoid(Re*gW0 + Im*gW1 + gb).
// One wave per 16-position tile; B fragment = wbar broadcast into all 16
// columns, so every column of D accumulates the same dot product.
// A-fragment layout (16-bit A 16x32, ISA 7.12.2):
//   lane l (m = l&15, hi = l>>4): halfs[0..7] = K[k0 + 8*hi .. +7],
//                                 halfs[8..15] = K[k0 + 16 + 8*hi .. +7]
// D layout (f32 16x16): lane l<16 vgpr r = D[r, l]; lane l>=16 = D[8+r, l-16].
// ---------------------------------------------------------------------------
__global__ __launch_bounds__(32) void bk_gate_wmma(
    const float* __restrict__ x, const _Float16* __restrict__ wbar,
    const float* __restrict__ bbar, const float* __restrict__ gW,
    const float* __restrict__ gb, float* __restrict__ gate)
{
  const int lane = threadIdx.x & 31;
  const int base = blockIdx.x << 4;           // 16 positions per wave
  const int m    = lane & 15;
  const int hi   = lane >> 4;
  const float*    xr = x    + (size_t)(base + m) * DD + (hi << 3);
  const _Float16* wr = wbar + (hi << 3);

  f8 c = {0.f, 0.f, 0.f, 0.f, 0.f, 0.f, 0.f, 0.f};

  for (int k0 = 0; k0 < DD; k0 += 32) {
    f4 x0 = *(const f4*)(xr + k0);
    f4 x1 = *(const f4*)(xr + k0 + 4);
    f4 x2 = *(const f4*)(xr + k0 + 16);
    f4 x3 = *(const f4*)(xr + k0 + 20);
    h8 w0 = *(const h8*)(wr + k0);
    h8 w1 = *(const h8*)(wr + k0 + 16);
    h16 a, b;
#pragma unroll
    for (int i = 0; i < 4; ++i) {
      a[i]      = (_Float16)x0[i];
      a[i + 4]  = (_Float16)x1[i];
      a[i + 8]  = (_Float16)x2[i];
      a[i + 12] = (_Float16)x3[i];
    }
#pragma unroll
    for (int i = 0; i < 8; ++i) {
      b[i]     = w0[i];
      b[i + 8] = w1[i];
    }
    // (neg_a, A, neg_b, B, c_mod, C, reuse_a, reuse_b)
    c = __builtin_amdgcn_wmma_f32_16x16x32_f16(false, a, false, b,
                                               (short)0, c, false, false);
  }

  // Extract: lanes 0..7 -> rows base+0..7 via c[l]; lanes 16..23 -> rows
  // base+8..15 via c[l-16] (all D columns equal, pick the diagonal lane).
  const bool lo = (lane < 8);
  const bool hi8 = (lane >= 16) && (lane < 24);
  if (lo || hi8) {
    const int r = lo ? lane : (lane - 8);
    const float he  = c[lane & 7] * (1.0f / 1024.0f) + bbar[0];
    const float aa  = he + 1e-6f;            // Re(he - z + EPS), z = 0 + 0.1i
    const float den = aa * aa + 0.01f;       // |he - z + EPS|^2
    const float re  = aa   / den;            // Re G
    const float im  = 0.1f / den;            // Im G
    const float targ = re * gW[0] + im * gW[1] + gb[0];
    gate[base + r] = 1.0f / (1.0f + expf(-targ));
  }
}

// ---------------------------------------------------------------------------
// Kernel 3 (the bandwidth kernel): one 256-thread block per 1024-float row of
// attention_weights. t = a * gate ; out = t / (sum_j t + EPS).
// 128 MB read + 128 MB write -> ~11 us at 23.3 TB/s.
//  - gate row (4KB) pulled into LDS via CDNA5 async DMA
//    (global_load_async_to_lds_b128, ASYNCcnt) -- each lane copies and
//    consumes only its own 16B, so a wave-level s_wait_asynccnt suffices.
//  - attention stream uses non-temporal load/store hints: in+out (256MB)
//    exceeds the 192MB L2, keep the streaming data from evicting the hot
//    gate/wbar tables.
// ---------------------------------------------------------------------------
__global__ __launch_bounds__(256) void bk_gated_norm(
    const float* __restrict__ attn, const float* __restrict__ gate,
    float* __restrict__ out)
{
  const int t = threadIdx.x;
  const size_t row = blockIdx.x;
  const int b = (int)(row / ROWS_PER_B);
  const float* arow = attn + (row << 10);
  const float* grow = gate + ((size_t)b << 10);

  __shared__ __align__(16) float sgate[SS];
  __shared__ float red[256];

  // async DMA: LDS[sgate + 16*t] = global[grow + 16*t]   (GVS addressing)
  {
    unsigned lds_addr = (unsigned)(uintptr_t)(&sgate[t << 2]);
    unsigned voff     = (unsigned)(t << 4);
    unsigned long long gbase = (unsigned long long)(uintptr_t)grow;
    asm volatile("global_load_async_to_lds_b128 %0, %1, %2"
                 :: "v"(lds_addr), "v"(voff), "s"(gbase)
                 : "memory");
  }

  // speculative prefetch of the next attention row (global_prefetch_b8)
  __builtin_prefetch(arow + SS + (t << 2), 0, 1);

  f4 a = __builtin_nontemporal_load((const f4*)arow + t);

  asm volatile("s_wait_asynccnt 0" ::: "memory");
  f4 g = *(const f4*)(sgate + (t << 2));

  f4 w = a * g;
  float p = (w[0] + w[1]) + (w[2] + w[3]);

  red[t] = p;
  __syncthreads();
  for (int off = 128; off > 0; off >>= 1) {
    if (t < off) red[t] += red[t + off];
    __syncthreads();
  }
  const float inv = 1.0f / (red[0] + 1e-6f);
  __builtin_nontemporal_store(w * inv, (f4*)(out + (row << 10)) + t);
}

// ---------------------------------------------------------------------------
// launch: inputs (x, attention_weights, Wd, bd, Wsup, bsup, Wsub, bsub, gW, gb)
// Wsup/bsup/Wsub/bsub are dead code in the reference -> never touched.
// Workspace layout:
//   [0,      4096)  wbar  (2048 f16, pre-scaled x1024)
//   [4096,   4100)  bbar  (f32)
//   [8192,  16384)  gate  (2048 f32)
//   [16384, 278528) column-sum partials (32 x 2048 f32)
// ---------------------------------------------------------------------------
extern "C" void kernel_launch(void* const* d_in, const int* in_sizes, int n_in,
                              void* d_out, int out_size, void* d_ws, size_t ws_size,
                              hipStream_t stream) {
  (void)in_sizes; (void)n_in; (void)out_size; (void)ws_size;
  const float* x    = (const float*)d_in[0];
  const float* attn = (const float*)d_in[1];
  const float* Wd   = (const float*)d_in[2];
  const float* bd   = (const float*)d_in[3];
  const float* gW   = (const float*)d_in[8];
  const float* gb   = (const float*)d_in[9];
  float* out = (float*)d_out;

  char* ws = (char*)d_ws;
  _Float16* wbar   = (_Float16*)(ws);
  float*    bbar   = (float*)(ws + 4096);
  float*    gate   = (float*)(ws + 8192);
  float*    partial= (float*)(ws + 16384);

  dim3 g1a(8, 32);
  bk_colsum_partial<<<g1a, 256, 0, stream>>>(Wd, partial);
  bk_finalize_wbar<<<9, 256, 0, stream>>>(partial, bd, wbar, bbar);
  bk_gate_wmma<<<PP / 16, 32, 0, stream>>>(x, wbar, bbar, gW, gb, gate);
  bk_gated_norm<<<NROWS, 256, 0, stream>>>(attn, gate, out);
}
